// scGO_7567732376246
// MI455X (gfx1250) — compile-verified, matrix operand order checked by reading, not run on hardware
//
#include <hip/hip_runtime.h>

// ---------------- problem constants ----------------
constexpr int B  = 2048;
constexpr int G  = 10000;
constexpr int T  = 1000;
constexpr int GO = 2000;
constexpr int C  = 32;
constexpr float NEG_SLOPE = 0.01f;

// padded dims: K pads to mult of 32, N pads to mult of 32 (zero rows)
constexpr int Gp  = 10016;  // 313*32
constexpr int Tp  = 1024;   // 32*32  (N-pad of layer1 == K-pad of layer2)
constexpr int GOp = 2016;   // 63*32  (N-pad of layer2/3 == K-pad of layer4)

// LDS tile: 32 rows x 32 K bf16, row stride padded to 40 u16 (80 B) so the
// 16 rows read by lanes 0-15 hit distinct banks (20*r mod 64 is a perm).
constexpr int STR = 40;
constexpr int TILE = 32 * STR;   // u16 elements per staged tile

// ---------------- workspace layout (u16 elements) ----------------
constexpr size_t XB_OFF   = 0;                                  // x bf16     [B,  Gp]
constexpr size_t G2T_OFF  = XB_OFF  + (size_t)B  * Gp;          // g2t^T bf16 [Tp, Gp]
constexpr size_t WM1_OFF  = G2T_OFF + (size_t)Tp * Gp;          // w1*m bf16  [Tp, Gp]
constexpr size_t WM2_OFF  = WM1_OFF + (size_t)Tp * Gp;          // w2*m bf16  [GOp,Tp]
constexpr size_t WM3_OFF  = WM2_OFF + (size_t)GOp * Tp;         // w3*m bf16  [GOp,Gp]
constexpr size_t WF2_OFF  = WM3_OFF + (size_t)GOp * Gp;         // fc2 bf16   [C,  GOp]
constexpr size_t TFS_OFF  = WF2_OFF + (size_t)C  * GOp;         // tf_sum bf16[B,  Tp]
constexpr size_t GOS_OFF  = TFS_OFF + (size_t)B  * Tp;          // go_sum bf16[B,  GOp]

// ---------------- types ----------------
typedef __attribute__((ext_vector_type(16))) __bf16 v16bf;
typedef __attribute__((ext_vector_type(8)))  float  v8f;

__device__ __forceinline__ unsigned short f2bf(float f) {
    // round-to-nearest-even f32 -> bf16
    unsigned u = __float_as_uint(f);
    u += 0x7FFFu + ((u >> 16) & 1u);
    return (unsigned short)(u >> 16);
}

__device__ __forceinline__ v8f wmma_bf16(v16bf a, v16bf b, v8f c) {
    return __builtin_amdgcn_wmma_f32_16x16x32_bf16(
        /*neg_a=*/false, a, /*neg_b=*/false, b,
        /*c_mod=*/(short)0, c, /*reuse_a=*/false, /*reuse_b=*/false);
}

// LDS byte offset of a generic pointer into __shared__: the LDS aperture maps
// addr[31:0] directly onto the wave's LDS allocation (ISA 10.2).
__device__ __forceinline__ unsigned lds_off(const void* p) {
    return (unsigned)(unsigned long long)p;
}

// Async DMA: copy 16 bytes per lane from global memory into LDS, bypassing
// VGPRs. Tracked by ASYNCcnt (GLOBAL_LOAD_ASYNC_TO_LDS_B128, VGLOBAL op 98).
__device__ __forceinline__ void async_cp16(unsigned lds_byte_off, const void* gaddr) {
    asm volatile("global_load_async_to_lds_b128 %0, %1, off"
                 :: "v"(lds_byte_off), "v"(gaddr)
                 : "memory");
}

// 16x32 bf16 fragment from a row-major [rows, ld] bf16 matrix in GLOBAL mem.
// CDNA5 16-bit operand layout: lane&15 -> row, lane>>4 -> K-half; each lane
// holds K blocks {h*8..h*8+7} and {16+h*8..+7} -> two global_load_b128.
__device__ __forceinline__ v16bf load_frag(const unsigned short* __restrict__ base,
                                           int ld, int row_base, int kbase, int lane) {
    const int row   = row_base + (lane & 15);
    const int khalf = (lane >> 4) * 8;
    const unsigned short* p = base + (size_t)row * ld + kbase + khalf;
    union { v16bf v; uint4 u[2]; } f;
    f.u[0] = *(const uint4*)(p);
    f.u[1] = *(const uint4*)(p + 16);
    return f.v;
}

// Same fragment shape but from a staged 32x32 LDS tile (row stride STR u16).
__device__ __forceinline__ v16bf lds_frag(const unsigned short* base,
                                          int row_base, int lane) {
    const int row   = row_base + (lane & 15);
    const int khalf = (lane >> 4) * 8;
    const unsigned short* p = base + row * STR + khalf;
    union { v16bf v; uint4 u[2]; } f;
    f.u[0] = *(const uint4*)(p);
    f.u[1] = *(const uint4*)(p + 16);
    return f.v;
}

// ---------------- pack kernels (fold static masks, cast to bf16, pad) ----
__global__ void __launch_bounds__(256)
pack_cast(const float* __restrict__ src, unsigned short* __restrict__ dst,
          int creal, int cpad, size_t total) {
    size_t i = (size_t)blockIdx.x * blockDim.x + threadIdx.x;
    if (i >= total) return;
    int r = (int)(i / cpad), c = (int)(i % cpad);
    float v = (c < creal) ? src[(size_t)r * creal + c] : 0.0f;
    dst[i] = f2bf(v);
}

__global__ void __launch_bounds__(256)
pack_mask(const float* __restrict__ w, const float* __restrict__ m,
          unsigned short* __restrict__ dst,
          int rreal, int creal, int cpad, size_t total) {
    size_t i = (size_t)blockIdx.x * blockDim.x + threadIdx.x;
    if (i >= total) return;
    int r = (int)(i / cpad), c = (int)(i % cpad);
    float v = 0.0f;
    if (r < rreal && c < creal) {
        size_t s = (size_t)r * creal + c;
        v = w[s] * m[s];
    }
    dst[i] = f2bf(v);
}

// g2t is [G, T] (K-major for our GEMM); transpose to [Tp, Gp] bf16
__global__ void __launch_bounds__(256)
pack_transpose(const float* __restrict__ src, unsigned short* __restrict__ dst,
               int rreal /*T*/, int creal /*G*/, int cpad /*Gp*/, size_t total) {
    size_t i = (size_t)blockIdx.x * blockDim.x + threadIdx.x;
    if (i >= total) return;
    int t = (int)(i / cpad), g = (int)(i % cpad);
    float v = (t < rreal && g < creal) ? src[(size_t)g * rreal + t] : 0.0f;
    dst[i] = f2bf(v);
}

// ---------------- layer 1: tf_res = x@g2t ; tf_fg = x@(w1*m)^T -----------
// Block: 8 waves, tile 128(M) x 32(N). Both 32x32 B tiles are DMA'd into
// double-buffered LDS via async copies (waves 0-3 stage g2t^T, waves 4-7
// stage w1*m) and shared by all 8 waves; A frags are direct global loads.
__global__ void __launch_bounds__(256)
gemm_l1(const unsigned short* __restrict__ xb,
        const unsigned short* __restrict__ wres,
        const unsigned short* __restrict__ wfg,
        float* __restrict__ out_tffg,          // [B, T]
        unsigned short* __restrict__ tfsum) {  // [B, Tp] bf16
    __shared__ unsigned short sR[2][TILE];
    __shared__ unsigned short sF[2][TILE];

    const int t    = threadIdx.x;
    const int lane = t & 31;
    const int wave = t >> 5;
    constexpr int TN = Tp / 32;               // 32 n-tiles
    const int tn   = blockIdx.x % TN;
    const int mblk = blockIdx.x / TN;
    const int mb   = mblk * 128 + wave * 16;
    const int nb   = tn * 32;

    // staging role (wave-uniform: waves 0-3 -> sR, waves 4-7 -> sF)
    const unsigned short* wsrc = (t < 128) ? wres : wfg;
    unsigned short (*sdst)[TILE] = (t < 128) ? sR : sF;
    const int sidx = t & 127;
    const int srow = sidx >> 2;               // 0..31
    const int scol = (sidx & 3) * 8;          // 0,8,16,24

    const unsigned short* gsrc = wsrc + (size_t)(nb + srow) * Gp + scol;
    const unsigned off0 = lds_off(&sdst[0][srow * STR + scol]);
    const unsigned off1 = lds_off(&sdst[1][srow * STR + scol]);

    constexpr int nK = Gp / 32;
    async_cp16(off0, gsrc);                   // prologue: buffer 0 <- k=0

    v8f accR[2] = {}, accF[2] = {};
    for (int kb = 0; kb < nK; ++kb) {
        const int cur = kb & 1;
        const int k   = kb * 32;
        if (kb + 1 < nK) {                    // prefetch next tile into other buf
            async_cp16(cur ? off0 : off1, gsrc + k + 32);
            asm volatile("s_wait_asynccnt 0x1" ::: "memory");  // oldest done
        } else {
            asm volatile("s_wait_asynccnt 0x0" ::: "memory");
        }
        __syncthreads();                      // all waves' tiles visible

        v16bf a = load_frag(xb, Gp, mb, k, lane);
#pragma unroll
        for (int s = 0; s < 2; ++s) {
            accR[s] = wmma_bf16(a, lds_frag(sR[cur], 16 * s, lane), accR[s]);
            accF[s] = wmma_bf16(a, lds_frag(sF[cur], 16 * s, lane), accF[s]);
        }
        __syncthreads();                      // done reading before overwrite
    }
    const int rb = mb + ((lane >> 4) * 8);
#pragma unroll
    for (int s = 0; s < 2; ++s) {
        const int col = nb + 16 * s + (lane & 15);
#pragma unroll
        for (int r = 0; r < 8; ++r) {
            float fg  = accF[s][r];
            float sum = accR[s][r] + fg;
            if (col < T) out_tffg[(size_t)(rb + r) * T + col] = fg;
            tfsum[(size_t)(rb + r) * Tp + col] = f2bf(sum);  // zero in pad cols
        }
    }
}

// ---------------- layers 2+3 fused: go_tf, go_sum ------------------------
// Block: 8 waves, tile 128(M) x 32(N). Double-buffered async-staged B tile
// reused across the two K loops (w2 over Tp, then w3 over Gp).
__global__ void __launch_bounds__(256)
gemm_l23(const unsigned short* __restrict__ tfsum,
         const unsigned short* __restrict__ w2,
         const unsigned short* __restrict__ xb,
         const unsigned short* __restrict__ w3,
         float* __restrict__ out_gotf,         // [B, GO]
         float* __restrict__ out_gosum,        // [B, GO]
         unsigned short* __restrict__ gos) {   // [B, GOp] bf16
    __shared__ unsigned short sB[2][TILE];

    const int t    = threadIdx.x;
    const int lane = t & 31;
    const int wave = t >> 5;
    constexpr int TN = GOp / 32;              // 63 n-tiles
    const int tn   = blockIdx.x % TN;
    const int mblk = blockIdx.x / TN;
    const int mb   = mblk * 128 + wave * 16;
    const int nb   = tn * 32;

    const int sidx = t & 127;
    const int srow = sidx >> 2;
    const int scol = (sidx & 3) * 8;
    const bool stager = (t < 128);            // wave-uniform

    const unsigned off0 = lds_off(&sB[0][srow * STR + scol]);
    const unsigned off1 = lds_off(&sB[1][srow * STR + scol]);
    const unsigned short* g2 = w2 + (size_t)(nb + srow) * Tp + scol;
    const unsigned short* g3 = w3 + (size_t)(nb + srow) * Gp + scol;

    v8f accA[2] = {}, accB[2] = {};

    constexpr int nK1 = Tp / 32;
    if (stager) async_cp16(off0, g2);
    for (int kb = 0; kb < nK1; ++kb) {        // go_from_tf: K = Tp
        const int cur = kb & 1;
        const int k   = kb * 32;
        if (kb + 1 < nK1) {
            if (stager) async_cp16(cur ? off0 : off1, g2 + k + 32);
            asm volatile("s_wait_asynccnt 0x1" ::: "memory");
        } else {
            asm volatile("s_wait_asynccnt 0x0" ::: "memory");
        }
        __syncthreads();
        v16bf a = load_frag(tfsum, Tp, mb, k, lane);
#pragma unroll
        for (int s = 0; s < 2; ++s)
            accA[s] = wmma_bf16(a, lds_frag(sB[cur], 16 * s, lane), accA[s]);
        __syncthreads();
    }

    constexpr int nK2 = Gp / 32;
    if (stager) async_cp16(off0, g3);
    for (int kb = 0; kb < nK2; ++kb) {        // go_from_gene: K = Gp
        const int cur = kb & 1;
        const int k   = kb * 32;
        if (kb + 1 < nK2) {
            if (stager) async_cp16(cur ? off0 : off1, g3 + k + 32);
            asm volatile("s_wait_asynccnt 0x1" ::: "memory");
        } else {
            asm volatile("s_wait_asynccnt 0x0" ::: "memory");
        }
        __syncthreads();
        v16bf a = load_frag(xb, Gp, mb, k, lane);
#pragma unroll
        for (int s = 0; s < 2; ++s)
            accB[s] = wmma_bf16(a, lds_frag(sB[cur], 16 * s, lane), accB[s]);
        __syncthreads();
    }

    const int rb = mb + ((lane >> 4) * 8);
#pragma unroll
    for (int s = 0; s < 2; ++s) {
        const int col = nb + 16 * s + (lane & 15);
#pragma unroll
        for (int r = 0; r < 8; ++r) {
            float gtf = accA[s][r];
            float sum = gtf + accB[s][r];
            float gs  = (sum > 0.0f) ? sum : sum * NEG_SLOPE;  // leaky relu
            if (col < GO) {
                out_gotf [(size_t)(rb + r) * GO + col] = gtf;
                out_gosum[(size_t)(rb + r) * GO + col] = gs;
            }
            gos[(size_t)(rb + r) * GOp + col] = f2bf(gs);     // zero in pad cols
        }
    }
}

// ---------------- layer 4: out = go_sum @ fc2^T (N = 32) -----------------
__global__ void __launch_bounds__(256)
gemm_l4(const unsigned short* __restrict__ gos,
        const unsigned short* __restrict__ wf2,
        float* __restrict__ out) {            // [B, C]
    __shared__ unsigned short sB[2][TILE];

    const int t    = threadIdx.x;
    const int lane = t & 31;
    const int wave = t >> 5;
    const int mb   = blockIdx.x * 128 + wave * 16;

    const int sidx = t & 127;
    const int srow = sidx >> 2;
    const int scol = (sidx & 3) * 8;
    const bool stager = (t < 128);

    const unsigned off0 = lds_off(&sB[0][srow * STR + scol]);
    const unsigned off1 = lds_off(&sB[1][srow * STR + scol]);
    const unsigned short* gw = wf2 + (size_t)srow * GOp + scol;

    constexpr int nK = GOp / 32;
    if (stager) async_cp16(off0, gw);

    v8f acc[2] = {};
    for (int kb = 0; kb < nK; ++kb) {
        const int cur = kb & 1;
        const int k   = kb * 32;
        if (kb + 1 < nK) {
            if (stager) async_cp16(cur ? off0 : off1, gw + k + 32);
            asm volatile("s_wait_asynccnt 0x1" ::: "memory");
        } else {
            asm volatile("s_wait_asynccnt 0x0" ::: "memory");
        }
        __syncthreads();
        v16bf a = load_frag(gos, GOp, mb, k, lane);
#pragma unroll
        for (int s = 0; s < 2; ++s)
            acc[s] = wmma_bf16(a, lds_frag(sB[cur], 16 * s, lane), acc[s]);
        __syncthreads();
    }
    const int rb = mb + ((lane >> 4) * 8);
#pragma unroll
    for (int s = 0; s < 2; ++s) {
        const int col = 16 * s + (lane & 15);
#pragma unroll
        for (int r = 0; r < 8; ++r)
            out[(size_t)(rb + r) * C + col] = acc[s][r];
    }
}

// ---------------- host launch --------------------------------------------
static inline unsigned cdiv(size_t n, unsigned b) { return (unsigned)((n + b - 1) / b); }

extern "C" void kernel_launch(void* const* d_in, const int* in_sizes, int n_in,
                              void* d_out, int out_size, void* d_ws, size_t ws_size,
                              hipStream_t stream) {
    const float* x    = (const float*)d_in[0];
    const float* g2t  = (const float*)d_in[1];
    const float* w1   = (const float*)d_in[2];  // [T, G]
    const float* w2   = (const float*)d_in[3];  // [GO, T]
    const float* w3   = (const float*)d_in[4];  // [GO, G]
    const float* wf2  = (const float*)d_in[5];  // [C, GO]
    const float* m1   = (const float*)d_in[6];
    const float* m2   = (const float*)d_in[7];
    const float* m3   = (const float*)d_in[8];
    const float* mf2  = (const float*)d_in[9];

    float* out       = (float*)d_out;                    // [B, C]
    float* out_gosum = out + (size_t)B * C;              // [B, GO]
    float* out_tffg  = out_gosum + (size_t)B * GO;       // [B, T]
    float* out_gotf  = out_tffg + (size_t)B * T;         // [B, GO]

    unsigned short* ws = (unsigned short*)d_ws;

    const unsigned TPB = 256;
    // pack: fold static masks, cast to bf16, zero-pad
    {
        size_t n;
        n = (size_t)B * Gp;
        pack_cast<<<cdiv(n, TPB), TPB, 0, stream>>>(x, ws + XB_OFF, G, Gp, n);
        n = (size_t)Tp * Gp;
        pack_transpose<<<cdiv(n, TPB), TPB, 0, stream>>>(g2t, ws + G2T_OFF, T, G, Gp, n);
        pack_mask<<<cdiv(n, TPB), TPB, 0, stream>>>(w1, m1, ws + WM1_OFF, T, G, Gp, n);
        n = (size_t)GOp * Tp;
        pack_mask<<<cdiv(n, TPB), TPB, 0, stream>>>(w2, m2, ws + WM2_OFF, GO, T, Tp, n);
        n = (size_t)GOp * Gp;
        pack_mask<<<cdiv(n, TPB), TPB, 0, stream>>>(w3, m3, ws + WM3_OFF, GO, G, Gp, n);
        n = (size_t)C * GOp;
        pack_mask<<<cdiv(n, TPB), TPB, 0, stream>>>(wf2, mf2, ws + WF2_OFF, C, GO, GOp, n);
    }

    // L1: 16 m-blocks * 32 n-tiles = 512 blocks (block tile 128x32)
    gemm_l1<<<dim3((B / 128) * (Tp / 32)), TPB, 0, stream>>>(
        ws + XB_OFF, ws + G2T_OFF, ws + WM1_OFF, out_tffg, ws + TFS_OFF);

    // L2+L3: 16 m-blocks * 63 n-tiles = 1008 blocks
    gemm_l23<<<dim3((B / 128) * (GOp / 32)), TPB, 0, stream>>>(
        ws + TFS_OFF, ws + WM2_OFF, ws + XB_OFF, ws + WM3_OFF,
        out_gotf, out_gosum, ws + GOS_OFF);

    // L4: 16 m-blocks (single 32-wide n-tile)
    gemm_l4<<<dim3(B / 128), TPB, 0, stream>>>(
        ws + GOS_OFF, ws + WF2_OFF, out);
}
